// AMKPDModel_Old_36240934043651
// MI455X (gfx1250) — matrix operand order repaired and checked by reading, not compiled, not run on hardware
//
#include <hip/hip_runtime.h>
#include <hip/hip_bf16.h>
#include <math.h>

#define BB 2
#define NN 1024
#define DD 512
#define HHD 8
#define DHD 64
#define LLY 4
#define INNER 1536
#define VOCABSZ 32000
#define LOOPSN 6
#define MTOK (BB*NN)   // 2048

typedef __attribute__((ext_vector_type(16))) __bf16          v16bf;
typedef __attribute__((ext_vector_type(16))) unsigned short  v16us;
typedef __attribute__((ext_vector_type(8)))  float           v8f;

__device__ __forceinline__ unsigned short f2bf(float f) {
    unsigned int u = __float_as_uint(f);
    u += 0x7FFFu + ((u >> 16) & 1u);      // round-to-nearest-even
    return (unsigned short)(u >> 16);
}

__device__ __forceinline__ v8f v8f_zero() {
    v8f z = {0.f,0.f,0.f,0.f,0.f,0.f,0.f,0.f};
    return z;
}

// load one 32-byte fragment (16 bf16) -- lowers to 2x b128 loads
__device__ __forceinline__ v16bf ldfrag(const unsigned short* p) {
    return __builtin_bit_cast(v16bf, *(const v16us*)p);
}

// ===========================================================================
// Repack kernels: f32 -> bf16 in WMMA fragment order (contiguous k = 16*half + slot, not the ISA table order)
// A frag (16x32):  lane = (m&15) + 16*k'[4],  slot = k'&15
// B frag (32x16):  lane = (n&15) + 16*k'[4],  slot = k'&15
// ===========================================================================
__global__ void repackA_kernel(const float* __restrict__ A, unsigned short* __restrict__ Af,
                               int M, int K)
{
    long long i = (long long)blockIdx.x * 256 + threadIdx.x;
    if (i >= (long long)M * K) return;
    int m = (int)(i / K), k = (int)(i % K);
    int mb = m >> 4, kb = k >> 5, kp = k & 31;
    int lane = (m & 15) + (((kp >> 4) & 1) << 4);
    int slot = kp & 15;
    int KB = K >> 5;
    Af[(((size_t)mb * KB + kb) * 32 + lane) * 16 + slot] = f2bf(A[i]);
}

__global__ void repackB_kernel(const float* __restrict__ W, unsigned short* __restrict__ Bf,
                               int K, int Ncol)
{
    long long i = (long long)blockIdx.x * 256 + threadIdx.x;
    if (i >= (long long)K * Ncol) return;
    int k = (int)(i / Ncol), n = (int)(i % Ncol);
    int kb = k >> 5, kp = k & 31;
    int nt = n >> 4;
    int lane = (n & 15) + (((kp >> 4) & 1) << 4);
    int slot = kp & 15;
    int NT = Ncol >> 4;
    Bf[(((size_t)kb * NT + nt) * 32 + lane) * 16 + slot] = f2bf(W[i]);
}

// phik -> transposed score-B fragments: per (b,h): [half(2)][keytile(64)][32][16]
__global__ void repack_phikT_kernel(const float* __restrict__ P, unsigned short* __restrict__ F)
{
    long long i = (long long)blockIdx.x * 256 + threadIdx.x;
    if (i >= (long long)MTOK * DD) return;
    int token = (int)(i / DD), c = (int)(i % DD);
    int b = token / NN, key = token % NN;
    int h = c / DHD, d = c % DHD;
    int half = d >> 5, kp = d & 31;
    int nt = key >> 4;
    int lane = (key & 15) + (((kp >> 4) & 1) << 4);
    int slot = kp & 15;
    size_t idx = ((((size_t)(b*HHD + h) * 2 + half) * (NN/16) + nt) * 32 + lane) * 16 + slot;
    F[idx] = f2bf(P[i]);
}

// v -> B fragments over keys: per (b,h): [keyblk(32)][dimtile(4)][32][16]
__global__ void repack_vB_kernel(const float* __restrict__ P, unsigned short* __restrict__ F)
{
    long long i = (long long)blockIdx.x * 256 + threadIdx.x;
    if (i >= (long long)MTOK * DD) return;
    int token = (int)(i / DD), c = (int)(i % DD);
    int b = token / NN, key = token % NN;
    int h = c / DHD, d = c % DHD;
    int kb = key >> 5, kp = key & 31;
    int nt = d >> 4;
    int lane = (d & 15) + (((kp >> 4) & 1) << 4);
    int slot = kp & 15;
    size_t idx = ((((size_t)(b*HHD + h) * 32 + kb) * 4 + nt) * 32 + lane) * 16 + slot;
    F[idx] = f2bf(P[i]);
}

// ===========================================================================
// GEMM on pre-packed fragments. 4 waves/block, 64x64 tile per block.
// B fragments (4KB/step, contiguous) are async-staged into a ping-pong LDS
// buffer one K-step ahead (global_load_async_to_lds_b128 + s_wait_asynccnt 2),
// shared by all 4 waves. epi: 0 = identity, 1 = elu(x)+1
// ===========================================================================
__global__ __launch_bounds__(128) void gemm_frag(
    const unsigned short* __restrict__ Af,   // [M/16][K/32][32][16]
    const unsigned short* __restrict__ Bf,   // [K/32][N/16][32][16]
    float* __restrict__ C, int M, int K, int Ncol, int epi, int accumulate)
{
    __shared__ v16us BshV[256];              // 2 x 4KB ping-pong stage
    unsigned short* Bsh = (unsigned short*)BshV;
    const int tid  = threadIdx.x;
    const int lane = tid & 31;
    const int wave = tid >> 5;
    const int mb   = blockIdx.y * 4 + wave;  // 16-row block
    const int nt0  = blockIdx.x * 4;         // 16-col subtile base
    const int KB = K >> 5;
    const int NT = Ncol >> 4;

    v8f acc[4];
#pragma unroll
    for (int t = 0; t < 4; ++t) acc[t] = v8f_zero();

    const unsigned ldsBase = (unsigned)(size_t)&Bsh[0];   // low 32 bits = LDS offset
    const unsigned tb = (unsigned)(tid * 32);             // this thread's 32B chunk

    auto issue = [&](int kb, int sel) {
        const unsigned short* src = Bf + ((size_t)kb * NT + nt0) * 512;
        unsigned lo = ldsBase + (unsigned)(sel * 4096) + tb;
        unsigned long long ga = (unsigned long long)(size_t)src + (unsigned long long)tb;
        asm volatile("global_load_async_to_lds_b128 %0, %1, off" :: "v"(lo), "v"(ga) : "memory");
        asm volatile("global_load_async_to_lds_b128 %0, %1, off" :: "v"(lo + 16u), "v"(ga + 16ull) : "memory");
    };

    issue(0, 0);
    for (int kb = 0; kb < KB; ++kb) {
        // A fragment load overlaps with the async wait
        v16bf a = ldfrag(Af + ((size_t)(mb * KB + kb) * 32 + lane) * 16);
        if (kb + 1 < KB) {
            issue(kb + 1, (kb + 1) & 1);     // stage next step into other buffer
            if (kb + 2 < KB)
                __builtin_prefetch(Bf + ((size_t)(kb + 2) * NT + nt0) * 512 + tid * 16, 0, 3);
            asm volatile("s_wait_asynccnt 0x2" ::: "memory");   // oldest batch (kb) done
        } else {
            asm volatile("s_wait_asynccnt 0x0" ::: "memory");
        }
        __syncthreads();

        const unsigned short* Bcur = Bsh + (kb & 1) * 2048;
        v16bf b0 = ldfrag(Bcur + (0*32 + lane) * 16);
        v16bf b1 = ldfrag(Bcur + (1*32 + lane) * 16);
        v16bf b2 = ldfrag(Bcur + (2*32 + lane) * 16);
        v16bf b3 = ldfrag(Bcur + (3*32 + lane) * 16);
        acc[0] = __builtin_amdgcn_wmma_f32_16x16x32_bf16(false, a, false, b0, (short)0, acc[0], false, false);
        acc[1] = __builtin_amdgcn_wmma_f32_16x16x32_bf16(false, a, false, b1, (short)0, acc[1], false, false);
        acc[2] = __builtin_amdgcn_wmma_f32_16x16x32_bf16(false, a, false, b2, (short)0, acc[2], false, false);
        acc[3] = __builtin_amdgcn_wmma_f32_16x16x32_bf16(false, a, false, b3, (short)0, acc[3], false, false);
        __syncthreads();                     // protect buffer for the kb+2 async
    }

#pragma unroll
    for (int t = 0; t < 4; ++t) {
#pragma unroll
        for (int r = 0; r < 8; ++r) {
            int row = mb * 16 + r + 8 * (lane >> 4);
            int col = (nt0 + t) * 16 + (lane & 15);
            size_t idx = (size_t)row * Ncol + col;
            float x = acc[t][r];
            if (epi == 1) x = (x > 0.f) ? (x + 1.f) : expf(x);   // elu(x)+1
            if (accumulate) x += C[idx];
            C[idx] = x;
        }
    }
}

// ===========================================================================
// Fused linear attention on pre-packed fragments.
// One wave per (b,h, 16-query tile). Per 32-key chunk: 4 clustered score-B
// loads -> 4 score WMMAs, fp32 relu^2 + norm accumulation, score tile stored
// straight into A-fragment order in LDS, 4 WMMAs against packed V fragments.
// ===========================================================================
__global__ __launch_bounds__(32) void attn_frag_kernel(
    const unsigned short* __restrict__ Aq,   // repackA(phiq): [mb(128)][kb(16)][32][16]
    const unsigned short* __restrict__ Kf,   // [bh][half2][nt64][32][16]
    const unsigned short* __restrict__ Vf,   // [bh][kb32][nt4][32][16]
    float* __restrict__ Cout)
{
    __shared__ v16us WfV[32];                // 1KB score-tile fragment buffer
    unsigned short* Wf = (unsigned short*)WfV;
    const int lane = threadIdx.x;
    const int qt = blockIdx.x;               // query tile 0..63
    const int bh = blockIdx.y;
    const int b = bh / HHD, h = bh % HHD;
    const int mb = b * (NN/16) + qt;         // row tile over MTOK/16
    const int KBD = DD >> 5;                 // 16

    const v16bf aq0 = ldfrag(Aq + ((size_t)(mb * KBD + h*2    ) * 32 + lane) * 16);
    const v16bf aq1 = ldfrag(Aq + ((size_t)(mb * KBD + h*2 + 1) * 32 + lane) * 16);

    const unsigned short* Kb_ = Kf + (size_t)bh * (2*64*512);
    const unsigned short* Vb_ = Vf + (size_t)bh * (32*4*512);

    v8f attr[4];
#pragma unroll
    for (int t = 0; t < 4; ++t) attr[t] = v8f_zero();
    float norm[8];
#pragma unroll
    for (int r = 0; r < 8; ++r) norm[r] = 0.f;
    const float scale = 0.125f;              // DH^-0.5

    for (int kb = 0; kb < NN/32; ++kb) {
        const int nt = kb * 2;
        // clustered score-B fragment loads (keys nt, nt+1; dim halves 0,1)
        v16bf bk00 = ldfrag(Kb_ + ((size_t)(0*64 + nt    ) * 32 + lane) * 16);
        v16bf bk01 = ldfrag(Kb_ + ((size_t)(1*64 + nt    ) * 32 + lane) * 16);
        v16bf bk10 = ldfrag(Kb_ + ((size_t)(0*64 + nt + 1) * 32 + lane) * 16);
        v16bf bk11 = ldfrag(Kb_ + ((size_t)(1*64 + nt + 1) * 32 + lane) * 16);
        v8f sc0 = v8f_zero(), sc1 = v8f_zero();
        sc0 = __builtin_amdgcn_wmma_f32_16x16x32_bf16(false, aq0, false, bk00, (short)0, sc0, false, false);
        sc0 = __builtin_amdgcn_wmma_f32_16x16x32_bf16(false, aq1, false, bk01, (short)0, sc0, false, false);
        sc1 = __builtin_amdgcn_wmma_f32_16x16x32_bf16(false, aq0, false, bk10, (short)0, sc1, false, false);
        sc1 = __builtin_amdgcn_wmma_f32_16x16x32_bf16(false, aq1, false, bk11, (short)0, sc1, false, false);

        // prefetch V fragments while doing the elementwise phase
        v16bf bv0 = ldfrag(Vb_ + ((size_t)(kb * 4 + 0) * 32 + lane) * 16);
        v16bf bv1 = ldfrag(Vb_ + ((size_t)(kb * 4 + 1) * 32 + lane) * 16);
        v16bf bv2 = ldfrag(Vb_ + ((size_t)(kb * 4 + 2) * 32 + lane) * 16);
        v16bf bv3 = ldfrag(Vb_ + ((size_t)(kb * 4 + 3) * 32 + lane) * 16);

#pragma unroll
        for (int s = 0; s < 2; ++s) {
#pragma unroll
            for (int r = 0; r < 8; ++r) {
                float w = (s == 0) ? sc0[r] : sc1[r];
                w *= scale;
                w = (w > 0.f) ? (w * w) : 0.f;
                norm[r] += w;
                int row = r + 8 * (lane >> 4);
                int kp  = 16*s + (lane & 15);          // key index within 32-chunk
                int laneW = (row & 15) + (((kp >> 4) & 1) << 4);
                int slotW = kp & 15;
                Wf[laneW * 16 + slotW] = f2bf(w);
            }
        }
        __syncthreads();
        v16bf aw = ldfrag(Wf + lane * 16);
        attr[0] = __builtin_amdgcn_wmma_f32_16x16x32_bf16(false, aw, false, bv0, (short)0, attr[0], false, false);
        attr[1] = __builtin_amdgcn_wmma_f32_16x16x32_bf16(false, aw, false, bv1, (short)0, attr[1], false, false);
        attr[2] = __builtin_amdgcn_wmma_f32_16x16x32_bf16(false, aw, false, bv2, (short)0, attr[2], false, false);
        attr[3] = __builtin_amdgcn_wmma_f32_16x16x32_bf16(false, aw, false, bv3, (short)0, attr[3], false, false);
        __syncthreads();
    }

#pragma unroll
    for (int r = 0; r < 8; ++r) {
        float v = norm[r];
        for (int off = 1; off < 16; off <<= 1) v += __shfl_xor(v, off, 32);
        norm[r] = v + 1e-6f;
    }
#pragma unroll
    for (int t = 0; t < 4; ++t) {
#pragma unroll
        for (int r = 0; r < 8; ++r) {
            int row = qt*16 + r + 8*(lane >> 4);
            int col = h*DHD + 16*t + (lane & 15);
            Cout[((size_t)b*NN + row) * DD + col] = attr[t][r] / norm[r];
        }
    }
}

// ===========================================================================
// Elementwise / norm / small kernels
// ===========================================================================
__global__ __launch_bounds__(128) void rmsadd_kernel(
    const float* __restrict__ a, const float* __restrict__ bsrc,
    const float* __restrict__ w, float* __restrict__ out, int hasB)
{
    __shared__ float warpsum[4];
    const int row = blockIdx.x;
    const int t = threadIdx.x;
    float x[4];
    float ss = 0.f;
#pragma unroll
    for (int i = 0; i < 4; ++i) {
        int c = t + 128*i;
        float v = a[(size_t)row*DD + c];
        if (hasB) v += bsrc[(size_t)row*DD + c];
        x[i] = v; ss += v*v;
    }
    for (int off = 16; off; off >>= 1) ss += __shfl_xor(ss, off, 32);
    if ((t & 31) == 0) warpsum[t >> 5] = ss;
    __syncthreads();
    float tot = warpsum[0] + warpsum[1] + warpsum[2] + warpsum[3];
    float r = rsqrtf(tot / (float)DD + 1.1920929e-07f);
#pragma unroll
    for (int i = 0; i < 4; ++i) {
        int c = t + 128*i;
        out[(size_t)row*DD + c] = x[i] * r * w[c];
    }
}

__global__ __launch_bounds__(128) void embed_init_kernel(
    const int* __restrict__ inputs, const int* __restrict__ carry_inputs,
    const unsigned char* __restrict__ carry_halted,
    const float* __restrict__ carry_hidden,
    const float* __restrict__ embedding, const float* __restrict__ pos_emb,
    const float* __restrict__ init_hidden,
    float* __restrict__ Xraw, float* __restrict__ Q)
{
    const int row = blockIdx.x;
    const int b = row / NN, n = row % NN;
    const int halt = carry_halted[b] != 0;
    const int tok = halt ? inputs[row] : carry_inputs[row];
    const int t = threadIdx.x;
#pragma unroll
    for (int i = 0; i < 4; ++i) {
        int c = t + 128*i;
        Xraw[(size_t)row*DD + c] = embedding[(size_t)tok*DD + c] + pos_emb[(size_t)n*DD + c];
        Q[(size_t)row*DD + c] = halt ? init_hidden[c] : carry_hidden[(size_t)row*DD + c];
    }
}

__global__ void axpy_kernel(const float* __restrict__ a, const float* __restrict__ b,
                            float* __restrict__ o, int n, float sb)
{
    int i = blockIdx.x * 256 + threadIdx.x;
    if (i < n) o[i] = a[i] + sb * b[i];
}

__global__ void swiglu_kernel(const float* __restrict__ GU, float* __restrict__ Hf)
{
    int i = blockIdx.x * 256 + threadIdx.x;
    if (i >= MTOK * INNER) return;
    int row = i / INNER, c = i % INNER;
    float g = GU[(size_t)row * (2*INNER) + c];
    float u = GU[(size_t)row * (2*INNER) + INNER + c];
    Hf[i] = (g / (1.f + expf(-g))) * u;
}

__global__ void conv_kernel(const float* __restrict__ Hf, const float* __restrict__ cw,
                            const float* __restrict__ cb, float* __restrict__ out)
{
    int i = blockIdx.x * 256 + threadIdx.x;
    if (i >= MTOK * INNER) return;
    int row = i / INNER, c = i % INNER;
    int b = row / NN, n = row % NN;
    float acc = cb[c];
#pragma unroll
    for (int j = 0; j < 3; ++j) {
        int nn = n + j - 1;
        if (nn >= 0 && nn < NN)
            acc += Hf[((size_t)b*NN + nn)*INNER + c] * cw[c*3 + j];
    }
    out[i] = acc / (1.f + expf(-acc));
}

__global__ void meanq_kernel(const float* __restrict__ Qn, float* __restrict__ meanv)
{
    int i = blockIdx.x * 256 + threadIdx.x;
    if (i >= BB * DD) return;
    int b = i / DD, d = i % DD;
    float s = 0.f;
    for (int n = 0; n < NN; ++n) s += Qn[((size_t)b*NN + n)*DD + d];
    meanv[i] = s / (float)NN;
}

__global__ void halt_kernel(const float* __restrict__ meanv,
                            const float* __restrict__ halt_w, const float* __restrict__ halt_b,
                            const int* __restrict__ steps, const unsigned char* __restrict__ halted_in,
                            float* __restrict__ outTail)
{
    int t = threadIdx.x;
    if (t < 4) {
        int b = t >> 1, j = t & 1;
        float s = halt_b[j];
        for (int d = 0; d < DD; ++d) s += meanv[b*DD + d] * halt_w[d*2 + j];
        outTail[(j == 0 ? 0 : 2) + b] = s;   // q_halt then q_cont
    }
    if (t < 2) {
        int st = (halted_in[t] ? 0 : steps[t]) + 1;
        outTail[4 + t] = (float)st;
        outTail[6 + t] = (st >= LOOPSN) ? 1.f : 0.f;
    }
}

// ===========================================================================
extern "C" void kernel_launch(void* const* d_in, const int* in_sizes, int n_in,
                              void* d_out, int out_size, void* d_ws, size_t ws_size,
                              hipStream_t stream)
{
    (void)in_sizes; (void)n_in; (void)out_size; (void)ws_size;
    const int*           inputs       = (const int*)d_in[0];
    const float*         carry_hidden = (const float*)d_in[2];
    const int*           carry_steps  = (const int*)d_in[3];
    const unsigned char* carry_halted = (const unsigned char*)d_in[4];
    const int*           carry_inputs = (const int*)d_in[5];
    const float* embedding    = (const float*)d_in[7];
    const float* pos_emb      = (const float*)d_in[8];
    const float* init_hidden  = (const float*)d_in[9];
    const float* input_norm_w = (const float*)d_in[10];
    const float* Wq    = (const float*)d_in[11];
    const float* Wk    = (const float*)d_in[12];
    const float* Wv    = (const float*)d_in[13];
    const float* Wo    = (const float*)d_in[14];
    const float* Waux  = (const float*)d_in[15];
    const float* Wup   = (const float*)d_in[16];
    const float* convw = (const float*)d_in[17];
    const float* convb = (const float*)d_in[18];
    const float* Wdown = (const float*)d_in[19];
    const float* norm1_w = (const float*)d_in[20];
    const float* norm2_w = (const float*)d_in[21];
    const float* final_norm_w = (const float*)d_in[22];
    const float* halt_w = (const float*)d_in[23];
    const float* halt_b = (const float*)d_in[24];
    const float* lm_head_w = (const float*)d_in[25];

    float* out = (float*)d_out;

    // ---- workspace carving (256B aligned) ----
    size_t off = 0;
    char* base = (char*)d_ws;
    auto alloc = [&](size_t bytes) -> void* {
        void* p = base + off;
        off += (bytes + 255) & ~(size_t)255;
        return p;
    };
    const size_t TD = (size_t)MTOK * DD;
    float* X     = (float*)alloc(TD * 4);
    float* Q     = (float*)alloc(TD * 4);
    float* Hc    = (float*)alloc(TD * 4);              // reused as m
    float* phiq  = (float*)alloc(TD * 4);              // reused as Hdown
    float* phik  = (float*)alloc(TD * 4);              // reused as Qn
    float* vb    = (float*)alloc(TD * 4);
    float* Cb    = (float*)alloc(TD * 4);
    float* mproj = (float*)alloc(TD * 4);
    float* Qi    = (float*)alloc(TD * 4);
    float* GU    = (float*)alloc((size_t)MTOK * 2 * INNER * 4);  // reused as conv out
    float* Hf    = (float*)alloc((size_t)MTOK * INNER * 4);
    float* meanv = (float*)alloc((size_t)BB * DD * 4);

    typedef unsigned short u16;
    const size_t WDD = (size_t)DD * DD;                          // 262144
    u16* WfQ    = (u16*)alloc(LLY * WDD * 2);
    u16* WfK    = (u16*)alloc(LLY * WDD * 2);
    u16* WfV    = (u16*)alloc(LLY * WDD * 2);
    u16* WfO    = (u16*)alloc(LLY * WDD * 2);
    u16* WfX    = (u16*)alloc(LLY * WDD * 2);
    u16* WfUp   = (u16*)alloc((size_t)LLY * DD * 2 * INNER * 2);
    u16* WfDown = (u16*)alloc((size_t)LLY * INNER * DD * 2);
    u16* WfHead = (u16*)alloc((size_t)DD * VOCABSZ * 2);
    u16* AfBig  = (u16*)alloc((size_t)MTOK * INNER * 2);         // activations, K<=1536
    u16* AqF    = (u16*)alloc(TD * 2);
    u16* KfF    = (u16*)alloc(TD * 2);
    u16* VfF    = (u16*)alloc(TD * 2);

    auto rpA = [&](const float* src, u16* dst, int M, int K) {
        long long n = (long long)M * K;
        repackA_kernel<<<(unsigned)((n + 255) / 256), 256, 0, stream>>>(src, dst, M, K);
    };
    auto rpB = [&](const float* src, u16* dst, int K, int Ncol) {
        long long n = (long long)K * Ncol;
        repackB_kernel<<<(unsigned)((n + 255) / 256), 256, 0, stream>>>(src, dst, K, Ncol);
    };
    auto gemm = [&](const u16* Af, const u16* Bf, float* C, int M, int K, int Ncol,
                    int epi, int acc) {
        gemm_frag<<<dim3(Ncol/64, M/64), 128, 0, stream>>>(Af, Bf, C, M, K, Ncol, epi, acc);
    };

    // ---- one-time weight repack (amortized over 8 block evaluations) ----
    for (int l = 0; l < LLY; ++l) {
        rpB(Wq   + l*WDD, WfQ + l*WDD, DD, DD);
        rpB(Wk   + l*WDD, WfK + l*WDD, DD, DD);
        rpB(Wv   + l*WDD, WfV + l*WDD, DD, DD);
        rpB(Wo   + l*WDD, WfO + l*WDD, DD, DD);
        rpB(Waux + l*WDD, WfX + l*WDD, DD, DD);
        rpB(Wup   + (size_t)l*DD*2*INNER, WfUp   + (size_t)l*DD*2*INNER, DD, 2*INNER);
        rpB(Wdown + (size_t)l*INNER*DD,   WfDown + (size_t)l*INNER*DD,   INNER, DD);
    }
    rpB(lm_head_w, WfHead, DD, VOCABSZ);

    const int elemBlocks = (MTOK*DD + 255) / 256;
    const int innerBlocks = (MTOK*INNER + 255) / 256;
    const int tdBlocks = (int)((TD + 255) / 256);

    embed_init_kernel<<<MTOK, 128, 0, stream>>>(inputs, carry_inputs, carry_halted,
        carry_hidden, embedding, pos_emb, init_hidden, X, Q);
    rmsadd_kernel<<<MTOK, 128, 0, stream>>>(X, nullptr, input_norm_w, X, 0);

    for (int it = 0; it < 2*LLY; ++it) {
        int l = it % LLY;
        axpy_kernel<<<elemBlocks, 256, 0, stream>>>(Q, X, Hc, MTOK*DD, 1.f);
        rpA(Hc, AfBig, MTOK, DD);
        gemm(AfBig, WfQ + l*WDD, phiq, MTOK, DD, DD, 1, 0);
        gemm(AfBig, WfK + l*WDD, phik, MTOK, DD, DD, 1, 0);
        gemm(AfBig, WfV + l*WDD, vb,   MTOK, DD, DD, 0, 0);

        rpA(phiq, AqF, MTOK, DD);
        repack_phikT_kernel<<<tdBlocks, 256, 0, stream>>>(phik, KfF);
        repack_vB_kernel<<<tdBlocks, 256, 0, stream>>>(vb, VfF);
        attn_frag_kernel<<<dim3(NN/16, BB*HHD), 32, 0, stream>>>(AqF, KfF, VfF, Cb);

        axpy_kernel<<<elemBlocks, 256, 0, stream>>>(Cb, vb, Hc, MTOK*DD, -1.f);  // m = C - v
        rpA(Hc, AfBig, MTOK, DD);
        gemm(AfBig, WfO + l*WDD, mproj, MTOK, DD, DD, 0, 0);
        rpA(Cb, AfBig, MTOK, DD);
        gemm(AfBig, WfX + l*WDD, mproj, MTOK, DD, DD, 0, 1);                     // +=

        rmsadd_kernel<<<MTOK, 128, 0, stream>>>(Q, mproj, norm1_w + (size_t)l*DD, Qi, 1);

        rpA(Qi, AfBig, MTOK, DD);
        gemm(AfBig, WfUp + (size_t)l*DD*2*INNER, GU, MTOK, DD, 2*INNER, 0, 0);
        swiglu_kernel<<<innerBlocks, 256, 0, stream>>>(GU, Hf);
        conv_kernel<<<innerBlocks, 256, 0, stream>>>(Hf, convw + (size_t)l*INNER*3,
                                                     convb + (size_t)l*INNER, GU);
        rpA(GU, AfBig, MTOK, INNER);
        gemm(AfBig, WfDown + (size_t)l*INNER*DD, phiq, MTOK, INNER, DD, 0, 0);   // Hdown
        rmsadd_kernel<<<MTOK, 128, 0, stream>>>(Qi, phiq, norm2_w + (size_t)l*DD, Q, 1);
    }

    rmsadd_kernel<<<MTOK, 128, 0, stream>>>(Q, nullptr, final_norm_w, phik, 0);  // Qn
    rpA(phik, AfBig, MTOK, DD);
    gemm(AfBig, WfHead, out, MTOK, DD, VOCABSZ, 0, 0);
    meanq_kernel<<<(BB*DD + 255)/256, 256, 0, stream>>>(phik, meanv);
    halt_kernel<<<1, 64, 0, stream>>>(meanv, halt_w, halt_b, carry_steps, carry_halted,
                                      out + (size_t)MTOK * VOCABSZ);
}
